// LSTMFinder_22866405884516
// MI455X (gfx1250) — matrix operand order repaired and checked by reading, not compiled, not run on hardware
//
#include <hip/hip_runtime.h>
#include <hip/hip_bf16.h>

typedef __attribute__((ext_vector_type(16))) _Float16 v16h;
typedef __attribute__((ext_vector_type(8)))  _Float16 v8h;
typedef __attribute__((ext_vector_type(8)))  float    v8f;

#define B_  256
#define W_  5
#define E_  200
#define D_  64
#define T_  5
#define K1  400   // 2E
#define K1P 416   // K1 padded to 32
#define K3  600   // 2E + E (x || h_parent)
#define K3P 608   // K3 padded to 32
#define NG  800   // 4E (gates)

#define KT1 (K1P / 32)          // 13 k-tiles for the MLP GEMMs
#define KT3 (K3P / 32)          // 19 k-tiles for the LSTM GEMM
#define NT1 (K1 / 16)           // 25 n-tiles (MLP)
#define NT3 (NG / 16)           // 50 n-tiles (LSTM)
#define SW1_HALVES (NT1 * KT1 * 512)   // 166400
#define SWL_HALVES (NT3 * KT3 * 512)   // 486400
#define OFF_W2 SW1_HALVES
#define OFF_WL (2 * SW1_HALVES)
#define TOTAL_HALVES (2 * SW1_HALVES + SWL_HALVES)  // 819200 (1.6 MB)

// ---- WMMA fragment loaders (layouts per CDNA5 ISA 7.12.2, wave32) ----

// A: 16x32 f16 tile from LDS row-major [16][lda]; two 16B vector loads.
// lanes 0-15: M=lane, K {k0+0..7, k0+16..23}; lanes 16-31: M=lane-16, +8.
__device__ __forceinline__ v16h load_a_frag(const _Float16* a, int lda, int k0) {
  int lane = threadIdx.x & 31;
  int m  = lane & 15;
  int hi = (lane >> 4) << 3;            // 0 or 8
  const _Float16* p = a + m * lda + k0 + hi;
  v8h lo  = *(const v8h*)(p);
  v8h hi8 = *(const v8h*)(p + 16);
  return __builtin_shufflevector(lo, hi8, 0, 1, 2, 3, 4, 5, 6, 7,
                                 8, 9, 10, 11, 12, 13, 14, 15);
}

// B: pre-swizzled fragment tiles: tile (nt,kt) at (nt*kTiles+kt)*512 halves,
// lane's 16 halves contiguous at lane*16. One 32B contiguous load per lane.
__device__ __forceinline__ v16h load_b_tile(const _Float16* tiles, int kTiles,
                                            int nt, int kt) {
  int lane = threadIdx.x & 31;
  const v8h* p = (const v8h*)(tiles + (((size_t)(nt * kTiles + kt)) << 9)
                                    + (lane << 4));
  v8h lo = p[0], hi = p[1];
  return __builtin_shufflevector(lo, hi, 0, 1, 2, 3, 4, 5, 6, 7,
                                 8, 9, 10, 11, 12, 13, 14, 15);
}

// ---- prep: pack f32 weights -> f16 B-fragment-swizzled workspace ----
// Fragment element map: within tile t=nt*KT+kt, lane l, elem e:
//   n = nt*16 + (l&15); k = kt*32 + ((l>>4)<<3) + (e&7) + ((e>>3)<<4); zero-pad k>=K.
__global__ void prep_kernel(const float* __restrict__ w1, const float* __restrict__ w2,
                            const float* __restrict__ wx, const float* __restrict__ wh,
                            _Float16* __restrict__ ws) {
  for (int i = blockIdx.x * blockDim.x + threadIdx.x; i < TOTAL_HALVES;
       i += gridDim.x * blockDim.x) {
    int seg, base;
    if (i < SW1_HALVES)            { seg = 0; base = i; }
    else if (i < 2 * SW1_HALVES)   { seg = 1; base = i - SW1_HALVES; }
    else                           { seg = 2; base = i - 2 * SW1_HALVES; }
    int kTiles = (seg == 2) ? KT3 : KT1;
    int N      = (seg == 2) ? NG  : K1;
    int K      = (seg == 2) ? K3  : K1;
    int tile = base >> 9;
    int rem  = base & 511;
    int lane = rem >> 4;
    int e    = rem & 15;
    int nt = tile / kTiles, kt = tile - nt * kTiles;
    int n = nt * 16 + (lane & 15);
    int k = kt * 32 + ((lane >> 4) << 3) + (e & 7) + ((e >> 3) << 4);
    float v = 0.f;
    if (k < K) {
      if (seg == 0)      v = w1[k * N + n];
      else if (seg == 1) v = w2[k * N + n];
      else               v = (k < K1) ? wx[k * NG + n] : wh[(k - K1) * NG + n];
    }
    ws[i] = (_Float16)v;
  }
}

__global__ __launch_bounds__(256)
void lstm_finder_kernel(const int* __restrict__ neighbor_ent,
                        const int* __restrict__ neighbor_rel,
                        const int* __restrict__ start,
                        const int* __restrict__ target,
                        const float* __restrict__ ent_emb,
                        const float* __restrict__ rel_emb,
                        const float* __restrict__ lstm_b,
                        const float* __restrict__ mlp_b1,
                        const float* __restrict__ mlp_b2,
                        const _Float16* __restrict__ wk,
                        float* __restrict__ out_h,
                        float* __restrict__ out_cur) {
  const _Float16* mw1 = wk;            // swizzled [25 x 13] tiles
  const _Float16* mw2 = wk + OFF_W2;   // swizzled [25 x 13] tiles
  const _Float16* wl  = wk + OFF_WL;   // swizzled [50 x 19] tiles

  __shared__ __align__(16) _Float16 a_h[16 * K3P];   // A staging (state_feat / x||ph)
  __shared__ __align__(16) _Float16 x_h[16 * K1P];   // MLP hidden (f16)
  __shared__ __align__(16) float c_s[W_][E_], h_s[W_][E_];
  __shared__ __align__(16) float pc_s[W_][E_], ph_s[W_][E_];
  __shared__ __align__(16) float feat_s[W_][K1];
  __shared__ __align__(16) float gates_s[W_][NG];
  __shared__ float scores_s[W_][D_];
  __shared__ int nb_e_s[W_][D_], nb_r_s[W_][D_];
  __shared__ int cur_s[W_];
  __shared__ int pw_s[W_], sel_r[W_], sel_e[W_], newcur_s[W_], pdone_s[W_];

  const int b   = blockIdx.x;
  const int tid = threadIdx.x;
  const int wv  = tid >> 5;               // wave id, wave32
  const int tgt = target[b];

  for (int t = tid; t < W_ * E_; t += 256) {
    c_s[t / E_][t % E_] = 0.f;
    h_s[t / E_][t % E_] = 0.f;
  }
  if (tid < W_) cur_s[tid] = start[b];
  __syncthreads();

  for (int step = 0; step < T_; ++step) {
    // ---- Phase 1: gather neighbors; build state_feat A tile; clear x_h ----
    for (int t = tid; t < 16 * K3P; t += 256) {
      int row = t / K3P, k = t % K3P;
      float v = 0.f;
      if (row < W_ && k < K1) v = (k < E_) ? c_s[row][k] : h_s[row][k - E_];
      a_h[t] = (_Float16)v;
    }
    for (int t = tid; t < 16 * K1P; t += 256) x_h[t] = (_Float16)0.f;
    for (int t = tid; t < W_ * D_; t += 256) {
      int w = t >> 6, d = t & 63;
      int cu = cur_s[w];
      nb_e_s[w][d] = neighbor_ent[(size_t)cu * D_ + d];
      nb_r_s[w][d] = neighbor_rel[(size_t)cu * D_ + d];
    }
    __syncthreads();

    // ---- GEMM1: relu(state_feat @ mlp_w1 + b1) -> x_h (f16) ----
    for (int nt = wv; nt < NT1; nt += 8) {
      v8f acc = {};
#pragma unroll 4
      for (int kt = 0; kt < KT1; ++kt) {
        v16h af = load_a_frag(a_h, K3P, kt * 32);
        v16h bf = load_b_tile(mw1, KT1, nt, kt);
        acc = __builtin_amdgcn_wmma_f32_16x16x32_f16(false, af, false, bf,
                                                     (short)0, acc, false, false);
      }
      int lane = tid & 31;
      int n  = nt * 16 + (lane & 15);
      int mb = (lane >> 4) << 3;
      float bias = mlp_b1[n];
#pragma unroll
      for (int r = 0; r < 8; ++r) {
        int m = mb + r;
        if (m < W_) {
          float v = acc[r] + bias;
          x_h[m * K1P + n] = (_Float16)(v > 0.f ? v : 0.f);
        }
      }
    }
    __syncthreads();

    // ---- GEMM2: relu(x @ mlp_w2 + b2) -> feat_s (f32) ----
    for (int nt = wv; nt < NT1; nt += 8) {
      v8f acc = {};
#pragma unroll 4
      for (int kt = 0; kt < KT1; ++kt) {
        v16h af = load_a_frag(x_h, K1P, kt * 32);
        v16h bf = load_b_tile(mw2, KT1, nt, kt);
        acc = __builtin_amdgcn_wmma_f32_16x16x32_f16(false, af, false, bf,
                                                     (short)0, acc, false, false);
      }
      int lane = tid & 31;
      int n  = nt * 16 + (lane & 15);
      int mb = (lane >> 4) << 3;
      float bias = mlp_b2[n];
#pragma unroll
      for (int r = 0; r < 8; ++r) {
        int m = mb + r;
        if (m < W_) {
          float v = acc[r] + bias;
          feat_s[m][n] = v > 0.f ? v : 0.f;
        }
      }
    }
    __syncthreads();

    // ---- scores: 320 gather-dots (L2-resident embedding rows) ----
    for (int t = tid; t < W_ * D_; t += 256) {
      int w = t >> 6, d = t & 63;
      const float4* rp = (const float4*)(rel_emb + (size_t)nb_r_s[w][d] * E_);
      const float4* ep = (const float4*)(ent_emb + (size_t)nb_e_s[w][d] * E_);
      const float4* f0 = (const float4*)(feat_s[w]);
      const float4* f1 = (const float4*)(feat_s[w] + E_);
      float s = 0.f;
#pragma unroll 4
      for (int k = 0; k < E_ / 4; ++k) {
        float4 a = rp[k], fb = f0[k];
        s += a.x * fb.x + a.y * fb.y + a.z * fb.z + a.w * fb.w;
      }
#pragma unroll 4
      for (int k = 0; k < E_ / 4; ++k) {
        float4 a = ep[k], fb = f1[k];
        s += a.x * fb.x + a.y * fb.y + a.z * fb.z + a.w * fb.w;
      }
      scores_s[w][d] = s;
    }
    __syncthreads();

    // ---- per-beam softmax + done/stay masking (wave w handles beam w) ----
    if (wv < W_) {
      int lane = tid & 31;
      float s0 = scores_s[wv][lane];
      float s1 = scores_s[wv][lane + 32];
      float m = fmaxf(s0, s1);
#pragma unroll
      for (int off = 16; off > 0; off >>= 1) m = fmaxf(m, __shfl_xor(m, off, 32));
      float e0 = __expf(s0 - m), e1 = __expf(s1 - m);
      float sum = e0 + e1;
#pragma unroll
      for (int off = 16; off > 0; off >>= 1) sum += __shfl_xor(sum, off, 32);
      float inv = 1.f / sum;
      float p0 = e0 * inv, p1 = e1 * inv;
      if (cur_s[wv] == tgt) { p0 = (lane == 0) ? 2.f : 0.f; p1 = 0.f; }
      scores_s[wv][lane]      = p0;
      scores_s[wv][lane + 32] = p1;
    }
    __syncthreads();

    // ---- top-W over the 320 probs (ties -> lowest index, matches top_k) ----
    if (tid == 0) {
      float* p = &scores_s[0][0];
      for (int j = 0; j < W_; ++j) {
        float best = -1.f; int bi = 0;
        for (int i = 0; i < W_ * D_; ++i)
          if (p[i] > best) { best = p[i]; bi = i; }
        p[bi] = -2.f;
        int pw = bi >> 6, d = bi & 63;
        pw_s[j] = pw;
        int pd = (cur_s[pw] == tgt);
        pdone_s[j]  = pd;
        sel_r[j]    = nb_r_s[pw][d];
        sel_e[j]    = nb_e_s[pw][d];
        newcur_s[j] = pd ? cur_s[pw] : nb_e_s[pw][d];
      }
    }
    __syncthreads();

    // ---- gather parent states; build A = [x || ph] for the LSTM GEMM ----
    for (int t = tid; t < W_ * E_; t += 256) {
      int j = t / E_, e = t % E_;
      pc_s[j][e] = c_s[pw_s[j]][e];
      ph_s[j][e] = h_s[pw_s[j]][e];
    }
    __syncthreads();
    for (int t = tid; t < W_ * K3; t += 256) {
      int j = t / K3, k = t % K3;
      float v;
      if (k < E_)          v = rel_emb[(size_t)sel_r[j] * E_ + k];
      else if (k < 2 * E_) v = ent_emb[(size_t)sel_e[j] * E_ + (k - E_)];
      else                 v = ph_s[j][k - 2 * E_];
      a_h[j * K3P + k] = (_Float16)v;
    }
    __syncthreads();

    // ---- GEMM3: [x||ph] @ [wx;wh] + b -> gates (f32) ----
    for (int nt = wv; nt < NT3; nt += 8) {
      v8f acc = {};
#pragma unroll 4
      for (int kt = 0; kt < KT3; ++kt) {
        v16h af = load_a_frag(a_h, K3P, kt * 32);
        v16h bf = load_b_tile(wl, KT3, nt, kt);
        acc = __builtin_amdgcn_wmma_f32_16x16x32_f16(false, af, false, bf,
                                                     (short)0, acc, false, false);
      }
      int lane = tid & 31;
      int n  = nt * 16 + (lane & 15);
      int mb = (lane >> 4) << 3;
      float bias = lstm_b[n];
#pragma unroll
      for (int r = 0; r < 8; ++r) {
        int m = mb + r;
        if (m < W_) gates_s[m][n] = acc[r] + bias;
      }
    }
    __syncthreads();

    // ---- LSTM elementwise update (i,f,o,g split order per jnp.split) ----
    for (int t = tid; t < W_ * E_; t += 256) {
      int j = t / E_, e = t % E_;
      float ig = gates_s[j][e];
      float fg = gates_s[j][E_ + e];
      float og = gates_s[j][2 * E_ + e];
      float gg = gates_s[j][3 * E_ + e];
      float pc = pc_s[j][e], ph = ph_s[j][e];
      float si = 1.f / (1.f + __expf(-ig));
      float sf = 1.f / (1.f + __expf(-fg));
      float so = 1.f / (1.f + __expf(-og));
      float nc = sf * pc + si * tanhf(gg);
      float nh = so * tanhf(nc);
      if (pdone_s[j]) { nc = pc; nh = ph; }
      c_s[j][e] = nc;
      h_s[j][e] = nh;
    }
    if (tid < W_) cur_s[tid] = newcur_s[tid];
    __syncthreads();
  }

  // ---- outputs: h [B,W,E] then cur [B,W] ----
  for (int t = tid; t < W_ * E_; t += 256)
    out_h[(size_t)b * (W_ * E_) + t] = h_s[t / E_][t % E_];
  if (tid < W_) out_cur[(size_t)b * W_ + tid] = (float)cur_s[tid];
}

extern "C" void kernel_launch(void* const* d_in, const int* in_sizes, int n_in,
                              void* d_out, int out_size, void* d_ws, size_t ws_size,
                              hipStream_t stream) {
  const int*   neighbor_ent = (const int*)d_in[0];
  const int*   neighbor_rel = (const int*)d_in[1];
  const int*   start        = (const int*)d_in[2];
  const int*   target       = (const int*)d_in[3];
  const float* ent_emb      = (const float*)d_in[4];
  const float* rel_emb      = (const float*)d_in[5];
  const float* lstm_wx      = (const float*)d_in[6];
  const float* lstm_wh      = (const float*)d_in[7];
  const float* lstm_b       = (const float*)d_in[8];
  const float* mlp_w1       = (const float*)d_in[9];
  const float* mlp_b1       = (const float*)d_in[10];
  const float* mlp_w2       = (const float*)d_in[11];
  const float* mlp_b2       = (const float*)d_in[12];

  _Float16* wk = (_Float16*)d_ws;   // 819200 halves = 1.6 MB, B-fragment order
  prep_kernel<<<800, 256, 0, stream>>>(mlp_w1, mlp_w2, lstm_wx, lstm_wh, wk);

  float* out_h   = (float*)d_out;
  float* out_cur = out_h + (size_t)B_ * W_ * E_;
  lstm_finder_kernel<<<B_, 256, 0, stream>>>(neighbor_ent, neighbor_rel, start, target,
                                             ent_emb, rel_emb, lstm_b, mlp_b1, mlp_b2,
                                             wk, out_h, out_cur);
}
